// GNNFingerprint3D_70377334112807
// MI455X (gfx1250) — matrix operand order repaired and checked by reading, not compiled, not run on hardware
//
#include <hip/hip_runtime.h>

typedef __attribute__((ext_vector_type(16))) _Float16 v16h;
typedef __attribute__((ext_vector_type(8)))  _Float16 v8h;
typedef __attribute__((ext_vector_type(4)))  _Float16 v4h;
typedef __attribute__((ext_vector_type(8)))  float    v8f;
typedef __attribute__((ext_vector_type(4)))  float    v4f;

#define L_ 6

// ---------------------------------------------------------------------------
// WMMA fragment loaders (ISA 7.12.2 layouts, wave32) — branch-free, vectorized
// ---------------------------------------------------------------------------

// A fragment 16x32 f16 from LDS (row-major, stride lda halfs).
// lane l: m=l&15, h=l>>4; elems 0..7 -> K=k0+8h+i ; 8..15 -> K=k0+16+8h+(i-8)
__device__ __forceinline__ v16h lds_load_a(const _Float16* __restrict__ sm,
                                           int lane, int lda, int k0) {
  int m = lane & 15, hh = lane >> 4;
  const _Float16* r = sm + m * lda + k0 + 8 * hh;
  v8h lo = *(const v8h*)(r);        // 16B aligned
  v8h hi = *(const v8h*)(r + 16);   // 16B aligned
  return __builtin_shufflevector(lo, hi, 0, 1, 2, 3, 4, 5, 6, 7,
                                 8, 9, 10, 11, 12, 13, 14, 15);
}

// A fragment 16x32 from global f32 rows (stride lda floats), cvt to f16.
__device__ __forceinline__ v16h glb_load_a32(const float* __restrict__ A,
                                             int lane, int lda, int k0) {
  int m = lane & 15, hh = lane >> 4;
  const float* r = A + (size_t)m * lda + k0 + 8 * hh;
  v4f f0 = *(const v4f*)(r);
  v4f f1 = *(const v4f*)(r + 4);
  v4f f2 = *(const v4f*)(r + 16);
  v4f f3 = *(const v4f*)(r + 20);
  v16h a;
#pragma unroll
  for (int i = 0; i < 4; ++i) {
    a[i]      = (_Float16)f0[i];
    a[4 + i]  = (_Float16)f1[i];
    a[8 + i]  = (_Float16)f2[i];
    a[12 + i] = (_Float16)f3[i];
  }
  return a;
}

// B fragment 32x16 from pre-transposed f16 weights Wt[N][K] (K = row stride).
// lane l: n=l&15, h=l>>4; elems 0..7 -> K=k0+8h+i ; 8..15 -> K=k0+16+8h+(i-8)
template <int K>
__device__ __forceinline__ v16h glb_load_bT(const _Float16* __restrict__ Wt,
                                            int lane, int k0, int n0) {
  int n = lane & 15, hh = lane >> 4;
  const _Float16* r = Wt + (size_t)(n0 + n) * K + k0 + 8 * hh;
  v8h lo = *(const v8h*)(r);        // 16B aligned
  v8h hi = *(const v8h*)(r + 16);   // 16B aligned
  return __builtin_shufflevector(lo, hi, 0, 1, 2, 3, 4, 5, 6, 7,
                                 8, 9, 10, 11, 12, 13, 14, 15);
}

__device__ __forceinline__ v8f wmma_f16(v16h a, v16h b, v8f c) {
  return __builtin_amdgcn_wmma_f32_16x16x32_f16(false, a, false, b,
                                                (short)0, c, false, false);
}

// Wave-scope GEMM: [16 x K] (LDS f16) x Wt[N][K] + bias -> LDS f16 [16 x N]
template <int K, int N, bool RELU>
__device__ __forceinline__ void wave_mlp_lds(const _Float16* __restrict__ Asm, int lda,
                                             const _Float16* __restrict__ Wt,
                                             const float* __restrict__ bias,
                                             _Float16* __restrict__ Osm, int lane) {
  int n = lane & 15, mh = lane >> 4;
  for (int n0 = 0; n0 < N; n0 += 16) {
    float bv = bias[n0 + n];
    v8f acc;
#pragma unroll
    for (int j = 0; j < 8; ++j) acc[j] = bv;
#pragma unroll
    for (int k0 = 0; k0 < K; k0 += 32) {
      v16h a = lds_load_a(Asm, lane, lda, k0);
      v16h b = glb_load_bT<K>(Wt, lane, k0, n0);
      acc = wmma_f16(a, b, acc);
    }
#pragma unroll
    for (int j = 0; j < 8; ++j) {
      float v = acc[j];
      if (RELU) v = fmaxf(v, 0.0f);
      Osm[(j + 8 * mh) * N + n0 + n] = (_Float16)v;
    }
  }
}

// ---------------------------------------------------------------------------
// Generic WMMA GEMM: one wave per 16-row tile. A [M x K] f32 contiguous,
// Wt [N x K] f16 transposed. SCATTER -> atomicAdd(out[srows[m]*N+n]).
// ---------------------------------------------------------------------------
template <int K, int N, bool SCATTER, bool RELU>
__global__ __launch_bounds__(32) void gemm16(const float* __restrict__ A,
                                             const _Float16* __restrict__ Wt,
                                             const float* __restrict__ bias,
                                             float* __restrict__ out,
                                             const int* __restrict__ srows) {
  int lane = threadIdx.x;
  int tile = blockIdx.x;
  const float* At = A + (size_t)tile * 16 * K;
  int n = lane & 15, mh = lane >> 4;
  for (int n0 = 0; n0 < N; n0 += 16) {
    float bv = bias[n0 + n];
    v8f acc;
#pragma unroll
    for (int j = 0; j < 8; ++j) acc[j] = bv;
#pragma unroll
    for (int k0 = 0; k0 < K; k0 += 32) {
      v16h a = glb_load_a32(At, lane, K, k0);
      v16h b = glb_load_bT<K>(Wt, lane, k0, n0);
      acc = wmma_f16(a, b, acc);
    }
#pragma unroll
    for (int j = 0; j < 8; ++j) {
      float v = acc[j];
      if (RELU) v = fmaxf(v, 0.0f);
      int m = tile * 16 + j + 8 * mh;
      if (SCATTER) atomicAdd(&out[(size_t)srows[m] * N + n0 + n], v);
      else         out[(size_t)m * N + n0 + n] = v;
    }
  }
}

// ---------------------------------------------------------------------------
// Fused GNN layer: one wave per 16-edge tile; full edge-MLP chain in LDS.
// ---------------------------------------------------------------------------
__global__ __launch_bounds__(32) void edge_layer_kernel(
    const float* __restrict__ hIn, float* __restrict__ hOut,
    float* __restrict__ ea,
    const int* __restrict__ rowp, const int* __restrict__ colp,
    const _Float16* __restrict__ w1, const float* __restrict__ b1,
    const _Float16* __restrict__ w2, const float* __restrict__ b2,
    const _Float16* __restrict__ w3, const float* __restrict__ b3,
    const _Float16* __restrict__ v1, const float* __restrict__ c1,
    const _Float16* __restrict__ v2, const float* __restrict__ c2) {
  __shared__ _Float16 s_nf[16 * 96];    // [hr | ea]
  __shared__ _Float16 s_ef[16 * 160];   // [hr | hc | ea]
  __shared__ _Float16 s_buf1[16 * 256]; // dh1 / de1
  __shared__ _Float16 s_buf2[16 * 512]; // dh2
  __shared__ int s_row[16];
  __shared__ int s_col[16];

  int lane = threadIdx.x;
  int e0 = blockIdx.x * 16;

  if (lane < 16) {
    s_row[lane] = rowp[e0 + lane];
    s_col[lane] = colp[e0 + lane];
  }
  __syncthreads();

  // gather hr/hc (16x64 f32, as 16B quads) and ea (16x32) into LDS f16
#pragma unroll
  for (int it = 0; it < 8; ++it) {
    int idx = it * 32 + lane;                 // 256 quad tasks
    int e = idx >> 4, q = (idx & 15) * 4;
    v4f hv = *(const v4f*)(hIn + (size_t)s_row[e] * 64 + q);
    v4f cv = *(const v4f*)(hIn + (size_t)s_col[e] * 64 + q);
    v4h h4, c4;
#pragma unroll
    for (int i = 0; i < 4; ++i) { h4[i] = (_Float16)hv[i]; c4[i] = (_Float16)cv[i]; }
    *(v4h*)(s_nf + e * 96 + q) = h4;
    *(v4h*)(s_ef + e * 160 + q) = h4;
    *(v4h*)(s_ef + e * 160 + 64 + q) = c4;
  }
#pragma unroll
  for (int it = 0; it < 4; ++it) {
    int idx = it * 32 + lane;                 // 128 quad tasks
    int e = idx >> 3, q = (idx & 7) * 4;
    v4f av = *(const v4f*)(ea + (size_t)(e0 + e) * 32 + q);
    v4h a4;
#pragma unroll
    for (int i = 0; i < 4; ++i) a4[i] = (_Float16)av[i];
    *(v4h*)(s_nf + e * 96 + 64 + q) = a4;
    *(v4h*)(s_ef + e * 160 + 128 + q) = a4;
  }
  __syncthreads();

  int n = lane & 15, mh = lane >> 4;

  // dh1 = relu(nf @ w1 + b1)        [16x96]x[96x256]
  wave_mlp_lds<96, 256, true>(s_nf, 96, w1, b1, s_buf1, lane);
  __syncthreads();
  // dh2 = relu(dh1 @ w2 + b2)       [16x256]x[256x512]
  wave_mlp_lds<256, 512, true>(s_buf1, 256, w2, b2, s_buf2, lane);
  __syncthreads();
  // dh3 = dh2 @ w3 + b3 -> atomic scatter into hOut[row]
  for (int n0 = 0; n0 < 64; n0 += 16) {
    float bv = b3[n0 + n];
    v8f acc;
#pragma unroll
    for (int j = 0; j < 8; ++j) acc[j] = bv;
#pragma unroll
    for (int k0 = 0; k0 < 512; k0 += 32) {
      v16h a = lds_load_a(s_buf2, lane, 512, k0);
      v16h b = glb_load_bT<512>(w3, lane, k0, n0);
      acc = wmma_f16(a, b, acc);
    }
#pragma unroll
    for (int j = 0; j < 8; ++j) {
      int m = j + 8 * mh;
      atomicAdd(&hOut[(size_t)s_row[m] * 64 + n0 + n], acc[j]);
    }
  }
  __syncthreads();

  // de1 = relu(ef @ v1 + c1)        [16x160]x[160x256]  (reuse buf1)
  wave_mlp_lds<160, 256, true>(s_ef, 160, v1, c1, s_buf1, lane);
  __syncthreads();
  // de = de1 @ v2 + c2 ; ea += de (tile-exclusive rows)
  for (int n0 = 0; n0 < 32; n0 += 16) {
    float bv = c2[n0 + n];
    v8f acc;
#pragma unroll
    for (int j = 0; j < 8; ++j) acc[j] = bv;
#pragma unroll
    for (int k0 = 0; k0 < 256; k0 += 32) {
      v16h a = lds_load_a(s_buf1, lane, 256, k0);
      v16h b = glb_load_bT<256>(v2, lane, k0, n0);
      acc = wmma_f16(a, b, acc);
    }
#pragma unroll
    for (int j = 0; j < 8; ++j) {
      size_t idx = (size_t)(e0 + j + 8 * mh) * 32 + n0 + n;
      ea[idx] += acc[j];
    }
  }
}

// ---------------------------------------------------------------------------
// Weight transpose+convert: W[K][N] f32 -> Wt[N][Kpad] f16 (zero-padded K)
// ---------------------------------------------------------------------------
__global__ void w_to_f16T(const float* __restrict__ W, _Float16* __restrict__ Wt,
                          int K, int Kpad, int N) {
  int i = blockIdx.x * 256 + threadIdx.x;
  if (i < N * Kpad) {
    int nn = i / Kpad, k = i - nn * Kpad;
    Wt[i] = (k < K) ? (_Float16)W[(size_t)k * N + nn] : (_Float16)0.0f;
  }
}

__global__ void pad_edge_attr(const float* __restrict__ ea16,
                              float* __restrict__ eadp, int E) {
  int i = blockIdx.x * 256 + threadIdx.x;
  if (i < E * 32) {
    int e = i >> 5, k = i & 31;
    eadp[i] = (k < 16) ? ea16[e * 16 + k] : 0.0f;
  }
}

// ---------------------------------------------------------------------------
// Small scalar kernels
// ---------------------------------------------------------------------------
__global__ void fill_f32(float* p, float v, int n) {
  int i = blockIdx.x * 256 + threadIdx.x;
  if (i < n) p[i] = v;
}

__global__ void edge_cnt(const int* __restrict__ rowp, float* cnt, int E) {
  int i = blockIdx.x * 256 + threadIdx.x;
  if (i < E) atomicAdd(&cnt[rowp[i]], 1.0f);
}

__global__ void hp_fin(float* hp, const float* __restrict__ agg,
                       const float* __restrict__ cnt, int NN) {
  int i = blockIdx.x * 256 + threadIdx.x;
  if (i < NN * 256) hp[i] += agg[i] / fmaxf(cnt[i >> 8], 1.0f);
}

__global__ __launch_bounds__(256) void qproj(const float* __restrict__ query,
                                             const float* __restrict__ wq,
                                             const float* __restrict__ bq,
                                             float* qv) {
  __shared__ float sq[256];
  int t = threadIdx.x;
  sq[t] = query[t];
  __syncthreads();
  float acc = bq[t];
  for (int k = 0; k < 256; ++k) acc += sq[k] * wq[k * 256 + t];
  qv[t] = acc;
}

__device__ __forceinline__ void atomicMaxFloat(float* addr, float val) {
  int* ai = (int*)addr;
  int old = *ai;
  while (__int_as_float(old) < val) {
    int assumed = old;
    old = atomicCAS(ai, assumed, __float_as_int(val));
    if (old == assumed) break;
  }
}

__global__ __launch_bounds__(256) void att_scores(const float* __restrict__ kbuf,
                                                  const float* __restrict__ qv,
                                                  float* scores, float* smax, int NN) {
  __shared__ float sq[256];
  __shared__ float red[256];
  int t = threadIdx.x;
  sq[t] = qv[t];
  __syncthreads();
  int node = blockIdx.x * 256 + t;
  float sc[4];
#pragma unroll
  for (int h = 0; h < 4; ++h) {
    float acc = -3.0e38f;
    if (node < NN) {
      const float* kr = kbuf + (size_t)node * 256 + h * 64;
      acc = 0.0f;
      for (int d = 0; d < 64; ++d) acc += sq[h * 64 + d] * kr[d];
      acc *= 0.125f; // 1/sqrt(64)
      scores[h * NN + node] = acc;
    }
    sc[h] = acc;
  }
  for (int h = 0; h < 4; ++h) {
    red[t] = sc[h];
    __syncthreads();
    for (int s = 128; s > 0; s >>= 1) {
      if (t < s) red[t] = fmaxf(red[t], red[t + s]);
      __syncthreads();
    }
    if (t == 0) atomicMaxFloat(&smax[h], red[0]);
    __syncthreads();
  }
}

__global__ __launch_bounds__(256) void att_accum(const float* __restrict__ scores,
                                                 const float* __restrict__ vbuf,
                                                 const float* __restrict__ smax,
                                                 float* numer, float* denom, int NN) {
  __shared__ float wsh[4 * 256];
  int t = threadIdx.x;
  int node = blockIdx.x * 256 + t;
#pragma unroll
  for (int h = 0; h < 4; ++h) {
    float w = 0.0f;
    if (node < NN) w = expf(scores[h * NN + node] - smax[h]);
    wsh[h * 256 + t] = w;
  }
  __syncthreads();
  int base = blockIdx.x * 256;
  int lim = NN - base; if (lim > 256) lim = 256;
  const float* wrow = wsh + (t >> 6) * 256;
  float acc = 0.0f;
  for (int i = 0; i < lim; ++i) acc += wrow[i] * vbuf[(size_t)(base + i) * 256 + t];
  atomicAdd(&numer[t], acc);
  if (t < 4) {
    float s = 0.0f;
    for (int i = 0; i < 256; ++i) s += wsh[t * 256 + i];
    atomicAdd(&denom[t], s);
  }
}

__global__ __launch_bounds__(256) void oproj(const float* __restrict__ numer,
                                             const float* __restrict__ denom,
                                             const float* __restrict__ wo,
                                             const float* __restrict__ bo, float* o2) {
  __shared__ float so[256];
  int t = threadIdx.x;
  so[t] = numer[t] / denom[t >> 6];
  __syncthreads();
  float acc = bo[t];
  for (int k = 0; k < 256; ++k) acc += so[k] * wo[k * 256 + t];
  o2[t] = acc;
}

__global__ __launch_bounds__(256) void head1(const float* __restrict__ o2,
                                             const float* __restrict__ h1w,
                                             const float* __restrict__ h1b,
                                             const float* __restrict__ lng,
                                             const float* __restrict__ lnb, float* zn) {
  __shared__ float st[256];
  __shared__ float red[256];
  int t = threadIdx.x;
  st[t] = fmaxf(o2[t], 0.0f);
  __syncthreads();
  float acc = h1b[t];
  for (int k = 0; k < 256; ++k) acc += st[k] * h1w[k * 256 + t];
  float z = fmaxf(acc, 0.0f);
  red[t] = z;
  __syncthreads();
  for (int s = 128; s > 0; s >>= 1) {
    if (t < s) red[t] += red[t + s];
    __syncthreads();
  }
  float mu = red[0] * (1.0f / 256.0f);
  __syncthreads();
  float d = z - mu;
  red[t] = d * d;
  __syncthreads();
  for (int s = 128; s > 0; s >>= 1) {
    if (t < s) red[t] += red[t + s];
    __syncthreads();
  }
  float var = red[0] * (1.0f / 256.0f);
  zn[t] = d * rsqrtf(var + 1e-5f) * lng[t] + lnb[t];
}

__global__ __launch_bounds__(256) void headout(const float* __restrict__ zn,
                                               const float* __restrict__ h2w,
                                               const float* __restrict__ h2b,
                                               float* out) {
  __shared__ float sz[256];
  sz[threadIdx.x] = zn[threadIdx.x];
  __syncthreads();
  int j = blockIdx.x * 256 + threadIdx.x;
  float acc = h2b[j];
  for (int k = 0; k < 256; ++k) acc += sz[k] * h2w[k * 1024 + j];
  out[j] = acc;
}

// ---------------------------------------------------------------------------
extern "C" void kernel_launch(void* const* d_in, const int* in_sizes, int n_in,
                              void* d_out, int out_size, void* d_ws, size_t ws_size,
                              hipStream_t stream) {
  const float* x         = (const float*)d_in[0];
  const float* edge_attr = (const float*)d_in[1];
  const int*   ei        = (const int*)d_in[2];
  const float* ne_w = (const float*)d_in[3];
  const float* ne_b = (const float*)d_in[4];
  const float* ee_w = (const float*)d_in[5];
  const float* ee_b = (const float*)d_in[6];
  const float* nw1 = (const float*)d_in[7];
  const float* nb1 = (const float*)d_in[8];
  const float* nw2 = (const float*)d_in[9];
  const float* nb2 = (const float*)d_in[10];
  const float* nw3 = (const float*)d_in[11];
  const float* nb3 = (const float*)d_in[12];
  const float* ew1 = (const float*)d_in[13];
  const float* eb1 = (const float*)d_in[14];
  const float* ew2 = (const float*)d_in[15];
  const float* eb2 = (const float*)d_in[16];
  const float* np_w = (const float*)d_in[17];
  const float* np_b = (const float*)d_in[18];
  const float* ep_w = (const float*)d_in[19];
  const float* ep_b = (const float*)d_in[20];
  const float* wq = (const float*)d_in[21];
  const float* bq = (const float*)d_in[22];
  const float* wk = (const float*)d_in[23];
  const float* bk = (const float*)d_in[24];
  const float* wv = (const float*)d_in[25];
  const float* bv = (const float*)d_in[26];
  const float* wo = (const float*)d_in[27];
  const float* bo = (const float*)d_in[28];
  const float* query = (const float*)d_in[29];
  const float* h1_w = (const float*)d_in[30];
  const float* h1_b = (const float*)d_in[31];
  const float* ln_g = (const float*)d_in[32];
  const float* ln_b = (const float*)d_in[33];
  const float* h2_w = (const float*)d_in[34];
  const float* h2_b = (const float*)d_in[35];

  const int NN = in_sizes[0] / 128;   // 20000
  const int E  = in_sizes[2] / 2;     // 100000
  const int* rowp = ei;
  const int* colp = ei + E;

  // ---- float workspace ----
  float* ws = (float*)d_ws;
  float* hA     = ws; ws += (size_t)NN * 64;
  float* hB     = ws; ws += (size_t)NN * 64;
  float* ea     = ws; ws += (size_t)E * 32;
  float* eadp   = ws; ws += (size_t)E * 32;
  float* hp     = ws; ws += (size_t)NN * 256;
  float* agg    = ws; ws += (size_t)NN * 256;
  float* cnt    = ws; ws += NN;
  float* kbuf   = ws; ws += (size_t)NN * 256;
  float* vbuf   = ws; ws += (size_t)NN * 256;
  float* scores = ws; ws += (size_t)4 * NN;
  float* smax   = ws; ws += 8;
  float* denom  = ws; ws += 8;
  float* numer  = ws; ws += 256;
  float* qv     = ws; ws += 256;
  float* o2     = ws; ws += 256;
  float* zn     = ws; ws += 256;

  // ---- f16 transposed-weight workspace (32B aligned: all counts mult of 16) ----
  _Float16* wt = (_Float16*)ws;
  _Float16* t_ne  = wt; wt += 64 * 128;
  _Float16* t_ee  = wt; wt += 32 * 32;
  _Float16* t_nw1 = wt; wt += (size_t)L_ * 256 * 96;
  _Float16* t_nw2 = wt; wt += (size_t)L_ * 512 * 256;
  _Float16* t_nw3 = wt; wt += (size_t)L_ * 64 * 512;
  _Float16* t_ew1 = wt; wt += (size_t)L_ * 256 * 160;
  _Float16* t_ew2 = wt; wt += (size_t)L_ * 32 * 256;
  _Float16* t_np  = wt; wt += 256 * 64;
  _Float16* t_ep  = wt; wt += 256 * 32;
  _Float16* t_wk  = wt; wt += 256 * 256;
  _Float16* t_wv  = wt; wt += 256 * 256;

  auto cvt = [&](const float* W, _Float16* Wt, int K, int Kpad, int N) {
    int total = N * Kpad;
    w_to_f16T<<<(total + 255) / 256, 256, 0, stream>>>(W, Wt, K, Kpad, N);
  };

  cvt(ne_w, t_ne, 128, 128, 64);
  cvt(ee_w, t_ee, 16, 32, 32);            // pad K 16 -> 32
  for (int l = 0; l < L_; ++l) {
    cvt(nw1 + (size_t)l * 96 * 256,  t_nw1 + (size_t)l * 256 * 96,  96, 96, 256);
    cvt(nw2 + (size_t)l * 256 * 512, t_nw2 + (size_t)l * 512 * 256, 256, 256, 512);
    cvt(nw3 + (size_t)l * 512 * 64,  t_nw3 + (size_t)l * 64 * 512,  512, 512, 64);
    cvt(ew1 + (size_t)l * 160 * 256, t_ew1 + (size_t)l * 256 * 160, 160, 160, 256);
    cvt(ew2 + (size_t)l * 256 * 32,  t_ew2 + (size_t)l * 32 * 256,  256, 256, 32);
  }
  cvt(np_w, t_np, 64, 64, 256);
  cvt(ep_w, t_ep, 32, 32, 256);
  cvt(wk, t_wk, 256, 256, 256);
  cvt(wv, t_wv, 256, 256, 256);

  // ---- embeddings ----
  gemm16<128, 64, false, false><<<NN / 16, 32, 0, stream>>>(x, t_ne, ne_b, hA, nullptr);
  pad_edge_attr<<<(E * 32 + 255) / 256, 256, 0, stream>>>(edge_attr, eadp, E);
  gemm16<32, 32, false, false><<<E / 16, 32, 0, stream>>>(eadp, t_ee, ee_b, ea, nullptr);

  // ---- GNN layers (ping-pong; seed next = current, atomic-add deltas) ----
  float* hCur = hA;
  float* hNxt = hB;
  for (int l = 0; l < L_; ++l) {
    hipMemcpyAsync(hNxt, hCur, (size_t)NN * 64 * sizeof(float),
                   hipMemcpyDeviceToDevice, stream);
    edge_layer_kernel<<<E / 16, 32, 0, stream>>>(
        hCur, hNxt, ea, rowp, colp,
        t_nw1 + (size_t)l * 256 * 96,  nb1 + (size_t)l * 256,
        t_nw2 + (size_t)l * 512 * 256, nb2 + (size_t)l * 512,
        t_nw3 + (size_t)l * 64 * 512,  nb3 + (size_t)l * 64,
        t_ew1 + (size_t)l * 256 * 160, eb1 + (size_t)l * 256,
        t_ew2 + (size_t)l * 32 * 256,  eb2 + (size_t)l * 32);
    float* t = hCur; hCur = hNxt; hNxt = t;
  }

  // ---- attention pooling ----
  fill_f32<<<(NN * 256 + 255) / 256, 256, 0, stream>>>(agg, 0.0f, NN * 256);
  fill_f32<<<(NN + 255) / 256, 256, 0, stream>>>(cnt, 0.0f, NN);
  gemm16<64, 256, false, false><<<NN / 16, 32, 0, stream>>>(hCur, t_np, np_b, hp, nullptr);
  gemm16<32, 256, true, false><<<E / 16, 32, 0, stream>>>(ea, t_ep, ep_b, agg, rowp);
  edge_cnt<<<(E + 255) / 256, 256, 0, stream>>>(rowp, cnt, E);
  hp_fin<<<(NN * 256 + 255) / 256, 256, 0, stream>>>(hp, agg, cnt, NN);

  gemm16<256, 256, false, false><<<NN / 16, 32, 0, stream>>>(hp, t_wk, bk, kbuf, nullptr);
  gemm16<256, 256, false, false><<<NN / 16, 32, 0, stream>>>(hp, t_wv, bv, vbuf, nullptr);
  qproj<<<1, 256, 0, stream>>>(query, wq, bq, qv);

  fill_f32<<<1, 256, 0, stream>>>(smax, -3.0e38f, 4);
  fill_f32<<<1, 256, 0, stream>>>(denom, 0.0f, 4);
  fill_f32<<<1, 256, 0, stream>>>(numer, 0.0f, 256);
  int nb = (NN + 255) / 256;
  att_scores<<<nb, 256, 0, stream>>>(kbuf, qv, scores, smax, NN);
  att_accum<<<nb, 256, 0, stream>>>(scores, vbuf, smax, numer, denom, NN);
  oproj<<<1, 256, 0, stream>>>(numer, denom, wo, bo, o2);

  // ---- projection head ----
  head1<<<1, 256, 0, stream>>>(o2, h1_w, h1_b, ln_g, ln_b, zn);
  headout<<<4, 256, 0, stream>>>(zn, h2_w, h2_b, (float*)d_out);
}
